// JumpProcess_12429635354694
// MI455X (gfx1250) — matrix operand order repaired and checked by reading, not compiled, not run on hardware
//
#include <hip/hip_runtime.h>
#include <hip/hip_bf16.h>
#include <math.h>

typedef __attribute__((ext_vector_type(16))) _Float16 v16h;
typedef __attribute__((ext_vector_type(8)))  float    v8f;

union V16H { v16h v; _Float16 h[16]; };
union V8F  { v8f  v; float    f[8];  };

static __device__ inline v8f wmma_f16(v16h a, v16h b, v8f c) {
  // D = A(16x32 f16) * B(32x16 f16) + C(16x16 f32)
  return __builtin_amdgcn_wmma_f32_16x16x32_f16(false, a, false, b, (short)0, c, false, false);
}

static __device__ inline float silu(float x) { return x / (1.0f + expf(-x)); }

// -------- A-operand element mapping (16-bit A, 16x32) --------
// lane l holds row m = l&15; half j (0..15): k = (j>>3)*16 + (l>>4)*8 + (j&7)
// -> two contiguous 8-element runs at kb and kb+16, kb = (l>>4)*8.
// -------- B-operand LDS staging (16-bit B, 32x16) --------
// lane l holds col n = l&15; half j: k = (j>>3)*16 + (l>>4)*8 + (j&7)  (same order as A)

// =====================================================================
// Kernel 1: protection net (D+1 -> 64 -> 1), writes jump_mask,
//           zeroes denom[] and added[], writes mask floats to d_out tail.
//           Persistent blocks: weights staged once, grid-stride tiles.
// =====================================================================
__global__ __launch_bounds__(256) void k_protect(
    const float* __restrict__ ns, const float* __restrict__ cap,
    const float* __restrict__ pw1, const float* __restrict__ pb1,
    const float* __restrict__ pw2, const float* __restrict__ pb2,
    int* __restrict__ jm, float* __restrict__ denom, float* __restrict__ added,
    float* __restrict__ maskOut, int N)
{
  // K padded to 5 chunks of 32 (129 valid)
  __shared__ _Float16 w1[5 * 4 * 32 * 16]; // [chunk][ntile][lane][16]
  __shared__ float b1[64], w2[64];
  const int tid = threadIdx.x;
  for (int idx = tid; idx < 5 * 4 * 32 * 16; idx += 256) {
    int j = idx & 15, l = (idx >> 4) & 31, ct = idx >> 9;
    int t = ct & 3, c = ct >> 2;
    int k = 32 * c + ((j >> 3) << 4) + ((l >> 4) << 3) + (j & 7);
    int n = 16 * t + (l & 15);
    w1[idx] = (_Float16)(k < 129 ? pw1[k * 64 + n] : 0.0f);
  }
  if (tid < 64) { b1[tid] = pb1[tid]; w2[tid] = pw2[tid]; }
  __syncthreads();

  const int wave = tid >> 5, l = tid & 31;
  const int numTiles = (N + 15) >> 4;
  const int waveStride = gridDim.x * 8;
  const float bias2 = pb2[0];

  for (int tile = blockIdx.x * 8 + wave; tile < numTiles; tile += waveStride) {
    int row = tile * 16 + (l & 15);
    if (row >= N) row = N - 1;
    const float* sp = ns + (size_t)row * 128;
    const float capv = cap[row];

    V8F acc[4];
    #pragma unroll
    for (int t = 0; t < 4; ++t) acc[t].v = (v8f)(0.0f);

    #pragma unroll
    for (int c = 0; c < 5; ++c) {
      V16H a;
      if (c < 4) {
        int kb = 32 * c + ((l >> 4) << 3);
        float4 x = *(const float4*)(sp + kb);
        float4 y = *(const float4*)(sp + kb + 4);
        float4 z = *(const float4*)(sp + kb + 16);
        float4 w = *(const float4*)(sp + kb + 20);
        a.h[0]=(_Float16)x.x; a.h[1]=(_Float16)x.y; a.h[2]=(_Float16)x.z; a.h[3]=(_Float16)x.w;
        a.h[4]=(_Float16)y.x; a.h[5]=(_Float16)y.y; a.h[6]=(_Float16)y.z; a.h[7]=(_Float16)y.w;
        a.h[8]=(_Float16)z.x; a.h[9]=(_Float16)z.y; a.h[10]=(_Float16)z.z; a.h[11]=(_Float16)z.w;
        a.h[12]=(_Float16)w.x; a.h[13]=(_Float16)w.y; a.h[14]=(_Float16)w.z; a.h[15]=(_Float16)w.w;
      } else {
        #pragma unroll
        for (int j = 0; j < 16; ++j) {
          int k = 128 + ((j >> 3) << 4) + ((l >> 4) << 3) + (j & 7);
          a.h[j] = (k == 128) ? (_Float16)capv : (_Float16)0.0f;
        }
      }
      #pragma unroll
      for (int t = 0; t < 4; ++t) {
        const v16h b = *(const v16h*)&w1[((c * 4 + t) * 32 + l) * 16];
        acc[t].v = wmma_f16(a.v, b, acc[t].v);
      }
    }

    float s[8];
    #pragma unroll
    for (int r = 0; r < 8; ++r) s[r] = 0.0f;
    #pragma unroll
    for (int t = 0; t < 4; ++t) {
      int n = 16 * t + (l & 15);
      float bb = b1[n], wn = w2[n];
      #pragma unroll
      for (int r = 0; r < 8; ++r) s[r] += silu(acc[t].f[r] + bb) * wn;
    }
    #pragma unroll
    for (int r = 0; r < 8; ++r)
      for (int off = 1; off < 16; off <<= 1) s[r] += __shfl_xor(s[r], off, 16);

    if ((l & 15) == 0) {
      #pragma unroll
      for (int r = 0; r < 8; ++r) {
        int node = tile * 16 + ((l >> 4) << 3) + r;
        if (node < N) {
          int mk = (s[r] + bias2) > 0.0f ? 1 : 0;
          jm[node] = mk;
          denom[node] = 0.0f;
          maskOut[node] = mk ? 1.0f : 0.0f;
        }
      }
    }
    // zero this tile's slice of added[]
    float* ap = added + (size_t)tile * 2048;
    size_t lim = (size_t)N * 128 - (size_t)tile * 2048;
    for (int i = l; i < 2048 && (size_t)i < lim; i += 32) ap[i] = 0.0f;
  }
}

// =====================================================================
// Kernel 2: edge MLP (256 -> 64 -> 64 -> 1), masked exp, denom atomics.
//           Persistent blocks amortize 40 KB LDS weight staging.
// =====================================================================
__global__ __launch_bounds__(256) void k_edge(
    const float* __restrict__ ns, const int* __restrict__ ei,
    const float* __restrict__ rw1, const float* __restrict__ rb1,
    const float* __restrict__ rw2, const float* __restrict__ rb2,
    const float* __restrict__ rw3, const float* __restrict__ rb3,
    const int* __restrict__ jm, float* __restrict__ eArr,
    float* __restrict__ denom, int N, int E)
{
  __shared__ _Float16 w1[8 * 4 * 32 * 16];   // 32 KB: layer1 B tiles
  __shared__ _Float16 w2[2 * 4 * 32 * 16];   // 8 KB:  layer2 B tiles
  __shared__ float b1[64], b2[64], w3[64];
  __shared__ _Float16 hbuf[8 * 16 * 64];     // 16 KB: per-wave hidden
  const int tid = threadIdx.x;
  for (int idx = tid; idx < 8 * 4 * 32 * 16; idx += 256) {
    int j = idx & 15, l = (idx >> 4) & 31, ct = idx >> 9;
    int t = ct & 3, c = ct >> 2;
    int k = 32 * c + ((j >> 3) << 4) + ((l >> 4) << 3) + (j & 7);
    int n = 16 * t + (l & 15);
    w1[idx] = (_Float16)rw1[k * 64 + n];
  }
  for (int idx = tid; idx < 2 * 4 * 32 * 16; idx += 256) {
    int j = idx & 15, l = (idx >> 4) & 31, ct = idx >> 9;
    int t = ct & 3, c = ct >> 2;
    int k = 32 * c + ((j >> 3) << 4) + ((l >> 4) << 3) + (j & 7);
    int n = 16 * t + (l & 15);
    w2[idx] = (_Float16)rw2[k * 64 + n];
  }
  if (tid < 64) { b1[tid] = rb1[tid]; b2[tid] = rb2[tid]; w3[tid] = rw3[tid]; }
  __syncthreads();

  const int wave = tid >> 5, l = tid & 31;
  const int numTiles = (E + 15) >> 4;
  const int waveStride = gridDim.x * 8;
  const int m = l & 15;
  const float b3 = rb3[0];
  _Float16* hb = &hbuf[wave * 1024];

  for (int tile = blockIdx.x * 8 + wave; tile < numTiles; tile += waveStride) {
    int e = tile * 16 + m; if (e >= E) e = E - 1;
    const int src = ei[e], tgt = ei[E + e];
    const float* sp = ns + (size_t)src * 128;
    const float* tp = ns + (size_t)tgt * 128;

    // ---- layer 1: 16x256 @ 256x64 ----
    V8F acc[4];
    #pragma unroll
    for (int t = 0; t < 4; ++t) acc[t].v = (v8f)(0.0f);

    #pragma unroll
    for (int c = 0; c < 8; ++c) {
      V16H a;
      int kb = 32 * c + ((l >> 4) << 3);
      {
        const float* p = (kb < 128) ? (sp + kb) : (tp + kb - 128);
        float4 x = *(const float4*)p, y = *(const float4*)(p + 4);
        a.h[0]=(_Float16)x.x; a.h[1]=(_Float16)x.y; a.h[2]=(_Float16)x.z; a.h[3]=(_Float16)x.w;
        a.h[4]=(_Float16)y.x; a.h[5]=(_Float16)y.y; a.h[6]=(_Float16)y.z; a.h[7]=(_Float16)y.w;
      }
      {
        int kc = kb + 16;
        const float* p = (kc < 128) ? (sp + kc) : (tp + kc - 128);
        float4 x = *(const float4*)p, y = *(const float4*)(p + 4);
        a.h[8]=(_Float16)x.x; a.h[9]=(_Float16)x.y; a.h[10]=(_Float16)x.z; a.h[11]=(_Float16)x.w;
        a.h[12]=(_Float16)y.x; a.h[13]=(_Float16)y.y; a.h[14]=(_Float16)y.z; a.h[15]=(_Float16)y.w;
      }
      #pragma unroll
      for (int t = 0; t < 4; ++t) {
        const v16h b = *(const v16h*)&w1[((c * 4 + t) * 32 + l) * 16];
        acc[t].v = wmma_f16(a.v, b, acc[t].v);
      }
    }

    // bias + SiLU, transpose C-layout -> row-major f16 in LDS
    #pragma unroll
    for (int t = 0; t < 4; ++t) {
      int n = 16 * t + (l & 15);
      float bb = b1[n];
      #pragma unroll
      for (int r = 0; r < 8; ++r) {
        int mm = r + ((l >> 4) << 3);
        hb[mm * 64 + n] = (_Float16)silu(acc[t].f[r] + bb);
      }
    }
    // same-wave LDS ops are in-order: safe to read back without barrier

    // ---- layer 2: 16x64 @ 64x64 ----
    V8F acc2[4];
    #pragma unroll
    for (int t = 0; t < 4; ++t) acc2[t].v = (v8f)(0.0f);
    #pragma unroll
    for (int c = 0; c < 2; ++c) {
      V16H a;
      #pragma unroll
      for (int j = 0; j < 16; ++j) {
        int k = 32 * c + ((j >> 3) << 4) + ((l >> 4) << 3) + (j & 7);
        a.h[j] = hb[m * 64 + k];
      }
      #pragma unroll
      for (int t = 0; t < 4; ++t) {
        const v16h b = *(const v16h*)&w2[((c * 4 + t) * 32 + l) * 16];
        acc2[t].v = wmma_f16(a.v, b, acc2[t].v);
      }
    }

    // ---- layer 3: dot with rw3, per-edge reduce ----
    float s[8];
    #pragma unroll
    for (int r = 0; r < 8; ++r) s[r] = 0.0f;
    #pragma unroll
    for (int t = 0; t < 4; ++t) {
      int n = 16 * t + (l & 15);
      float bb = b2[n], wn = w3[n];
      #pragma unroll
      for (int r = 0; r < 8; ++r) s[r] += silu(acc2[t].f[r] + bb) * wn;
    }
    #pragma unroll
    for (int r = 0; r < 8; ++r)
      for (int off = 1; off < 16; off <<= 1) s[r] += __shfl_xor(s[r], off, 16);

    if ((l & 15) == 0) {
      #pragma unroll
      for (int r = 0; r < 8; ++r) {
        int em = tile * 16 + ((l >> 4) << 3) + r;
        if (em < E) {
          int s2 = ei[em], t2 = ei[E + em];
          float raw = s[r] + b3;
          // softmax is shift-invariant: skip segment-max, use exp directly
          float ev = (jm[s2] && !jm[t2]) ? expf(raw) : 0.0f;
          eArr[em] = ev;
          if (ev > 0.0f) atomicAdd(&denom[s2], ev);
        }
      }
    }
  }
}

// =====================================================================
// Kernel 3: messages = (e/denom[src]) * states[src] scatter-added to tgt
// =====================================================================
__global__ __launch_bounds__(256) void k_scatter(
    const float* __restrict__ ns, const int* __restrict__ ei,
    const float* __restrict__ eArr, const float* __restrict__ denom,
    float* __restrict__ added, int E)
{
  const int wave = threadIdx.x >> 5, l = threadIdx.x & 31;
  const int stride = gridDim.x * 8;
  for (int e = blockIdx.x * 8 + wave; e < E; e += stride) {
    const float ev = eArr[e];
    if (ev <= 0.0f) continue;
    const int src = ei[e], tgt = ei[E + e];
    const float d = denom[src];
    const float w = ev / (d > 0.0f ? d : 1.0f);
    const float* sp = ns + (size_t)src * 128;
    float* ap = added + (size_t)tgt * 128;
    #pragma unroll
    for (int i = 0; i < 4; ++i) {
      int idx = l + 32 * i;
      atomicAdd(&ap[idx], w * sp[idx]);
    }
  }
}

// =====================================================================
// Kernel 4: redistributed = ns+added; tanh(redist @ tw + tb)*0.05 select
// =====================================================================
__global__ __launch_bounds__(256) void k_final(
    const float* __restrict__ ns, const float* __restrict__ added,
    const float* __restrict__ tw, const float* __restrict__ tb,
    const int* __restrict__ jm, float* __restrict__ out, int N)
{
  __shared__ _Float16 wt[4 * 8 * 32 * 16]; // 32 KB: [chunk][ntile][lane][16]
  __shared__ float bt[128];
  const int tid = threadIdx.x;
  for (int idx = tid; idx < 4 * 8 * 32 * 16; idx += 256) {
    int j = idx & 15, l = (idx >> 4) & 31, ct = idx >> 9;
    int t = ct & 7, c = ct >> 3;
    int k = 32 * c + ((j >> 3) << 4) + ((l >> 4) << 3) + (j & 7);
    int n = 16 * t + (l & 15);
    wt[idx] = (_Float16)tw[k * 128 + n];
  }
  if (tid < 128) bt[tid] = tb[tid];
  __syncthreads();

  const int wave = tid >> 5, l = tid & 31;
  const int numTiles = (N + 15) >> 4;
  const int waveStride = gridDim.x * 8;

  for (int tile = blockIdx.x * 8 + wave; tile < numTiles; tile += waveStride) {
    int row = tile * 16 + (l & 15);
    if (row >= N) row = N - 1;
    const float* sp = ns + (size_t)row * 128;
    const float* ap = added + (size_t)row * 128;

    V8F acc[8];
    #pragma unroll
    for (int t = 0; t < 8; ++t) acc[t].v = (v8f)(0.0f);

    #pragma unroll
    for (int c = 0; c < 4; ++c) {
      V16H a;
      int kb = 32 * c + ((l >> 4) << 3);
      #pragma unroll
      for (int j = 0; j < 8; ++j) a.h[j]     = (_Float16)(sp[kb + j]      + ap[kb + j]);
      #pragma unroll
      for (int j = 0; j < 8; ++j) a.h[8 + j] = (_Float16)(sp[kb + 16 + j] + ap[kb + 16 + j]);
      #pragma unroll
      for (int t = 0; t < 8; ++t) {
        const v16h b = *(const v16h*)&wt[((c * 8 + t) * 32 + l) * 16];
        acc[t].v = wmma_f16(a.v, b, acc[t].v);
      }
    }

    #pragma unroll
    for (int t = 0; t < 8; ++t) {
      int n = 16 * t + (l & 15);
      float bb = bt[n];
      #pragma unroll
      for (int r = 0; r < 8; ++r) {
        int node = tile * 16 + ((l >> 4) << 3) + r;
        if (node < N) {
          float prot = tanhf(acc[t].f[r] + bb) * 0.05f;
          float redist = ns[(size_t)node * 128 + n] + added[(size_t)node * 128 + n];
          out[(size_t)node * 128 + n] = jm[node] ? prot : redist;
        }
      }
    }
  }
}

// =====================================================================
extern "C" void kernel_launch(void* const* d_in, const int* in_sizes, int n_in,
                              void* d_out, int out_size, void* d_ws, size_t ws_size,
                              hipStream_t stream) {
  const float* ns   = (const float*)d_in[0];
  const float* cap  = (const float*)d_in[1];
  const int*   ei   = (const int*)d_in[2];   // JAX default x64-off -> int32
  const float* rw1  = (const float*)d_in[3];
  const float* rb1  = (const float*)d_in[4];
  const float* rw2  = (const float*)d_in[5];
  const float* rb2  = (const float*)d_in[6];
  const float* rw3  = (const float*)d_in[7];
  const float* rb3  = (const float*)d_in[8];
  const float* pw1  = (const float*)d_in[9];
  const float* pb1  = (const float*)d_in[10];
  const float* pw2  = (const float*)d_in[11];
  const float* pb2  = (const float*)d_in[12];
  const float* tw   = (const float*)d_in[13];
  const float* tb   = (const float*)d_in[14];

  const int N = in_sizes[1];
  const int E = in_sizes[2] / 2;

  auto align256 = [](size_t x) { return (x + 255) & ~(size_t)255; };
  char* w = (char*)d_ws;
  int*   jm    = (int*)w;                 w += align256((size_t)N * sizeof(int));
  float* denom = (float*)w;               w += align256((size_t)N * sizeof(float));
  float* eArr  = (float*)w;               w += align256((size_t)E * sizeof(float));
  float* added = (float*)w;               w += align256((size_t)N * 128 * sizeof(float));
  (void)ws_size; (void)n_in; (void)out_size;

  float* outStates = (float*)d_out;
  float* outMask   = outStates + (size_t)N * 128;

  const int tilesN = (N + 15) / 16;
  const int tilesE = (E + 15) / 16;

  // Persistent-block grids: enough waves to fill the device, small enough to
  // amortize per-block LDS weight staging across many tiles.
  int blocksA = (tilesN + 7) / 8; if (blocksA > 128) blocksA = 128;
  int blocksC = (tilesE + 7) / 8; if (blocksC > 512) blocksC = 512;
  int blocksD = (E + 7) / 8;      if (blocksD > 4096) blocksD = 4096;

  k_protect<<<blocksA, 256, 0, stream>>>(ns, cap, pw1, pb1, pw2, pb2,
                                         jm, denom, added, outMask, N);
  k_edge<<<blocksC, 256, 0, stream>>>(ns, ei, rw1, rb1, rw2, rb2, rw3, rb3,
                                      jm, eArr, denom, N, E);
  k_scatter<<<blocksD, 256, 0, stream>>>(ns, ei, eArr, denom, added, E);
  k_final<<<blocksA, 256, 0, stream>>>(ns, added, tw, tb, jm, outStates, N);
}
